// MultiHeadAttention_5970004541871
// MI455X (gfx1250) — matrix operand order repaired and checked by reading, not compiled, not run on hardware
//
#include <hip/hip_runtime.h>
#include <hip/hip_bf16.h>

typedef __attribute__((ext_vector_type(16))) __bf16 v16bf;
typedef __attribute__((ext_vector_type(8)))  __bf16 bf16x8;
typedef __attribute__((ext_vector_type(8)))  float  v8f;

#define EMBED 2048
#define SEQ   2048
#define BATCH 2
#define HEADS 16
#define HDIM  128
#define ROWS  (BATCH*SEQ)   /* 4096 */

__device__ __forceinline__ v16bf cat8(bf16x8 lo, bf16x8 hi) {
  return __builtin_shufflevector(lo, hi, 0,1,2,3,4,5,6,7,8,9,10,11,12,13,14,15);
}
__device__ __forceinline__ v8f wmma_bf16(v16bf a, v16bf b, v8f c) {
  // D = A(16x32 bf16) x B(32x16 bf16) + C(16x16 f32)
  return __builtin_amdgcn_wmma_f32_16x16x32_bf16(false, a, false, b, (short)0, c,
                                                 false, false);
}
__device__ __forceinline__ v8f vzero8() {
  v8f z;
#pragma unroll
  for (int i = 0; i < 8; ++i) z[i] = 0.0f;
  return z;
}

// ------------------------------------------------------- TDM (CDNA5) helper
#if defined(__has_builtin)
#if __has_builtin(__builtin_amdgcn_tensor_load_to_lds) && \
    __has_builtin(__builtin_amdgcn_s_wait_tensorcnt)
#define HAVE_TDM 1
#endif
#endif

#ifdef HAVE_TDM
typedef __attribute__((ext_vector_type(4))) unsigned int u32x4;
typedef __attribute__((ext_vector_type(4))) int          i32x4;
typedef __attribute__((ext_vector_type(8))) int          i32x8;

// 2D row-major tile DMA: global -> LDS via the Tensor Data Mover.
// D# bitfields per CDNA5 ISA ch.8 (groups 0/1; 2D so groups 2/3 are zero).
__device__ __forceinline__ void tdm_load_tile_2d(unsigned lds_byte_addr,
                                                 const void* global_tile,
                                                 unsigned d0_elems,  // contiguous
                                                 unsigned d1_rows,
                                                 unsigned row_stride_elems) {
  const unsigned long long ga = (unsigned long long)(uintptr_t)global_tile;
  u32x4 g0;
  g0[0] = 1u;                        // count=1 (valid user descriptor)
  g0[1] = lds_byte_addr;             // lds_addr
  g0[2] = (unsigned)ga;              // global_addr[31:0]
  g0[3] = (unsigned)((ga >> 32) & 0x01ffffffull) | 0x80000000u; // addr[56:32]|type=2
  i32x8 g1;
  g1[0] = 0x00010000;                               // data_size=1 (2 B/elem)
  g1[1] = (int)(d0_elems << 16);                    // tensor_dim0[15:0]
  g1[2] = (int)((d0_elems >> 16) | (d1_rows << 16));// dim0[31:16] | dim1[15:0]
  g1[3] = (int)((d1_rows >> 16) | (d0_elems << 16));// dim1[31:16] | tile_dim0
  g1[4] = (int)d1_rows;                             // tile_dim1 (tile_dim2=0)
  g1[5] = (int)row_stride_elems;                    // tensor_dim0_stride[31:0]
  g1[6] = 0;
  g1[7] = 0;
  i32x4 gz;
  gz[0] = 0; gz[1] = 0; gz[2] = 0; gz[3] = 0;
#if __clang_major__ >= 23
  i32x8 gz8;
#pragma unroll
  for (int i = 0; i < 8; ++i) gz8[i] = 0;
  __builtin_amdgcn_tensor_load_to_lds(g0, g1, gz, gz, gz8, 0);
#else
  __builtin_amdgcn_tensor_load_to_lds(g0, g1, gz, gz, 0);
#endif
}
#endif  // HAVE_TDM

// ---------------------------------------------------------------- convert
__global__ void f32_to_bf16_kernel(const float* __restrict__ in,
                                   __bf16* __restrict__ out, int n) {
  int i = blockIdx.x * blockDim.x + threadIdx.x;
  if (i < n) out[i] = (__bf16)in[i];
}

// ---------------------------------------------------------------- GEMM
// C[M,N] = A[M,K] * B[K,N], A/B bf16 row-major, C f32 or bf16.
// Block: 128x128 tile, BK=32, 256 threads = 8 waves; wave -> 32x64 patch.
// A tile staged by TDM (linear rows); B tile staged manually (transposed).
template <typename CT>
__global__ __launch_bounds__(256) void gemm_bf16_kernel(
    const __bf16* __restrict__ A, const __bf16* __restrict__ B,
    CT* __restrict__ C, int M, int N, int K) {
  __shared__ __bf16 As[128][32];  // [m][k]
  __shared__ __bf16 Bt[128][32];  // [n][k]  (B staged transposed)

  const int tid  = threadIdx.x;
  const int lane = tid & 31;
  const int wave = tid >> 5;      // 0..7
  const int wr   = wave >> 1;     // 0..3 -> M offset wr*32
  const int wc   = wave & 1;      // 0..1 -> N offset wc*64
  const int lrow = lane & 15;
  const int lhi  = lane >> 4;     // 0/1
  const int bm   = blockIdx.y * 128;
  const int bn   = blockIdx.x * 128;

  v8f acc[2][4];
#pragma unroll
  for (int m = 0; m < 2; ++m)
#pragma unroll
    for (int n = 0; n < 4; ++n) acc[m][n] = vzero8();

  const int ar = tid >> 1, ak = (tid & 1) * 16;   // A tile fallback mapping
  const int bk = tid >> 3, bnn = (tid & 7) * 16;  // B tile: 16 elems/thread
  const int kbA = lhi * 8;    // A-frag K base: lanes0-15 -> {0..7,16..23}
  const int kbB = lhi * 8;    // B-frag K base: lanes0-15 -> {0..7,16..23}

  for (int k0 = 0; k0 < K; k0 += 32) {
#ifdef HAVE_TDM
    if (wave == 0)
      tdm_load_tile_2d((unsigned)(uintptr_t)&As[0][0],
                       A + (size_t)bm * K + k0, /*d0=*/32, /*rows=*/128,
                       /*stride=*/(unsigned)K);
#else
    {
      const __bf16* ap = A + (size_t)(bm + ar) * K + k0 + ak;
      *(bf16x8*)&As[ar][ak]     = *(const bf16x8*)ap;
      *(bf16x8*)&As[ar][ak + 8] = *(const bf16x8*)(ap + 8);
    }
#endif
    const __bf16* bp = B + (size_t)(k0 + bk) * N + bn + bnn;
    bf16x8 b0 = *(const bf16x8*)bp;
    bf16x8 b1 = *(const bf16x8*)(bp + 8);
#pragma unroll
    for (int i = 0; i < 8; ++i) {
      Bt[bnn + i][bk]     = b0[i];
      Bt[bnn + 8 + i][bk] = b1[i];
    }
#ifdef HAVE_TDM
    if (wave == 0) __builtin_amdgcn_s_wait_tensorcnt((short)0);
#endif
    __syncthreads();

    v16bf af[2], bfr[4];
#pragma unroll
    for (int m = 0; m < 2; ++m) {
      const __bf16* p = &As[wr * 32 + m * 16 + lrow][kbA];
      af[m] = cat8(*(const bf16x8*)p, *(const bf16x8*)(p + 16));
    }
#pragma unroll
    for (int n = 0; n < 4; ++n) {
      const __bf16* p = &Bt[wc * 64 + n * 16 + lrow][kbB];
      bfr[n] = cat8(*(const bf16x8*)p, *(const bf16x8*)(p + 16));
    }
#pragma unroll
    for (int m = 0; m < 2; ++m)
#pragma unroll
      for (int n = 0; n < 4; ++n)
        acc[m][n] = wmma_bf16(af[m], bfr[n], acc[m][n]);
    __syncthreads();
  }

#pragma unroll
  for (int m = 0; m < 2; ++m)
#pragma unroll
    for (int n = 0; n < 4; ++n) {
      const int row0 = bm + wr * 32 + m * 16 + lhi * 8;
      const int col  = bn + wc * 64 + n * 16 + lrow;
#pragma unroll
      for (int r = 0; r < 8; ++r)
        C[(size_t)(row0 + r) * N + col] = (CT)acc[m][n][r];
    }
}

// ---------------------------------------------------------------- RoPE
// In-place rotary embedding on bf16 Q,K in [b, s, h*d] layout.
// One thread per (b, s, h, i) with i in 0..63 (rotation pair (i, i+64)).
__global__ void rope_inplace_kernel(__bf16* __restrict__ Q,
                                    __bf16* __restrict__ K) {
  const int idx = blockIdx.x * blockDim.x + threadIdx.x;
  const int i = idx & 63;
  const int h = (idx >> 6) & 15;
  const int s = (idx >> 10) & 2047;
  const int b = idx >> 21;
  const size_t base = (size_t)(b * SEQ + s) * EMBED + h * HDIM;
  // freq = 10000^(-2i/128) = exp2(-log2(10000) * 2i/128)
  const float freq = exp2f(-13.287712379549449f * (float)(2 * i) * (1.0f / 128.0f));
  const float ang = (float)s * freq;
  const float c = __cosf(ang), sn = __sinf(ang);
  {
    float x1 = (float)Q[base + i], x2 = (float)Q[base + 64 + i];
    Q[base + i]      = (__bf16)(x1 * c - x2 * sn);
    Q[base + 64 + i] = (__bf16)(x2 * c + x1 * sn);
  }
  {
    float x1 = (float)K[base + i], x2 = (float)K[base + 64 + i];
    K[base + i]      = (__bf16)(x1 * c - x2 * sn);
    K[base + 64 + i] = (__bf16)(x2 * c + x1 * sn);
  }
}

// ---------------------------------------------------------------- attention
// Causal flash attention. Q/K/V bf16 in [b, s, h*d]; O bf16 in [b, s, h*d].
// Grid: (s/128, b*h); 256 threads = 8 waves, wave w owns q rows [q0, q0+16).
// K tile staged by TDM; V tile transposed manually; next-V prefetched.
__global__ __launch_bounds__(256) void flash_attn_kernel(
    const __bf16* __restrict__ Q, const __bf16* __restrict__ K,
    const __bf16* __restrict__ V, __bf16* __restrict__ O) {
  __shared__ __bf16 Ks[32][128];      // key rows   [kv][d]
  __shared__ __bf16 Vt[128][32];      // value cols [d][kv]
  __shared__ __bf16 Ps[8][16][32];    // per-wave P relayout patch

  const int tid  = threadIdx.x;
  const int lane = tid & 31;
  const int wave = tid >> 5;
  const int lrow = lane & 15;
  const int lhi  = lane >> 4;
  const int bh   = blockIdx.y;
  const int b    = bh >> 4, h = bh & 15;
  const int qblk = blockIdx.x;
  const int q0   = qblk * 128 + wave * 16;
  const int akb  = lhi * 8;

  const __bf16* Qh = Q + (size_t)b * SEQ * EMBED + h * HDIM;
  const __bf16* Kh = K + (size_t)b * SEQ * EMBED + h * HDIM;
  const __bf16* Vh = V + (size_t)b * SEQ * EMBED + h * HDIM;

  // Q tile (16 x 128) as 4 A-fragments, kept in VGPRs for all key blocks.
  v16bf qf[4];
#pragma unroll
  for (int c = 0; c < 4; ++c) {
    const __bf16* p = Qh + (size_t)(q0 + lrow) * EMBED + c * 32 + akb;
    qf[c] = cat8(*(const bf16x8*)p, *(const bf16x8*)(p + 16));
  }

  v8f o[8];
#pragma unroll
  for (int n = 0; n < 8; ++n) o[n] = vzero8();
  float mrow[8], lsum[8];
#pragma unroll
  for (int r = 0; r < 8; ++r) { mrow[r] = -1e30f; lsum[r] = 0.0f; }

  const int kvr = tid >> 3;
  const int kvd = (tid & 7) * 16;
  const int kvend = (qblk + 1) * 128;

  for (int kv0 = 0; kv0 < kvend; kv0 += 32) {
    // ---- stage K (TDM) and V-transposed (manual) ----
#ifdef HAVE_TDM
    if (wave == 0)
      tdm_load_tile_2d((unsigned)(uintptr_t)&Ks[0][0],
                       Kh + (size_t)kv0 * EMBED, /*d0=*/128, /*rows=*/32,
                       /*stride=*/EMBED);
#else
    {
      const __bf16* kp = Kh + (size_t)(kv0 + kvr) * EMBED + kvd;
      *(bf16x8*)&Ks[kvr][kvd]     = *(const bf16x8*)kp;
      *(bf16x8*)&Ks[kvr][kvd + 8] = *(const bf16x8*)(kp + 8);
    }
#endif
    const __bf16* vp = Vh + (size_t)(kv0 + kvr) * EMBED + kvd;
    bf16x8 v0 = *(const bf16x8*)vp;
    bf16x8 v1 = *(const bf16x8*)(vp + 8);
#pragma unroll
    for (int i = 0; i < 8; ++i) {
      Vt[kvd + i][kvr]     = v0[i];
      Vt[kvd + 8 + i][kvr] = v1[i];
    }
    if (kv0 + 32 < kvend)  // hide next block's V behind this block's compute
      __builtin_prefetch(vp + 32 * EMBED, 0, 1);
#ifdef HAVE_TDM
    if (wave == 0) __builtin_amdgcn_s_wait_tensorcnt((short)0);
#endif
    __syncthreads();

    if (kv0 <= q0 + 15) {  // wave-uniform causal skip (EXEC stays all-ones)
      // S = Q * K^T for this 16q x 32k block (two 16x16 tiles).
      v8f s0 = vzero8(), s1 = vzero8();
#pragma unroll
      for (int c = 0; c < 4; ++c) {
        const __bf16* p0 = &Ks[lrow][c * 32 + lhi * 8];
        const __bf16* p1 = &Ks[16 + lrow][c * 32 + lhi * 8];
        v16bf kf0 = cat8(*(const bf16x8*)p0, *(const bf16x8*)(p0 + 16));
        v16bf kf1 = cat8(*(const bf16x8*)p1, *(const bf16x8*)(p1 + 16));
        s0 = wmma_bf16(qf[c], kf0, s0);
        s1 = wmma_bf16(qf[c], kf1, s1);
      }
      const float scale = 0.088388347648318447f;  // 1/sqrt(128)
#pragma unroll
      for (int r = 0; r < 8; ++r) {
        const int qr = q0 + lhi * 8 + r;
        float x0 = (kv0 + lrow      <= qr) ? s0[r] * scale : -1e30f;
        float x1 = (kv0 + 16 + lrow <= qr) ? s1[r] * scale : -1e30f;
        float mx = fmaxf(x0, x1);
        mx = fmaxf(mx, __shfl_xor(mx, 8));
        mx = fmaxf(mx, __shfl_xor(mx, 4));
        mx = fmaxf(mx, __shfl_xor(mx, 2));
        mx = fmaxf(mx, __shfl_xor(mx, 1));
        const float mnew = fmaxf(mrow[r], mx);
        const float corr = __expf(mrow[r] - mnew);
        const float p0 = __expf(x0 - mnew);
        const float p1 = __expf(x1 - mnew);
        float sum = p0 + p1;
        sum += __shfl_xor(sum, 8);
        sum += __shfl_xor(sum, 4);
        sum += __shfl_xor(sum, 2);
        sum += __shfl_xor(sum, 1);
        lsum[r] = lsum[r] * corr + sum;
        mrow[r] = mnew;
#pragma unroll
        for (int n = 0; n < 8; ++n) o[n][r] *= corr;
        Ps[wave][lhi * 8 + r][lrow]      = (__bf16)p0;   // C-layout -> LDS
        Ps[wave][lhi * 8 + r][16 + lrow] = (__bf16)p1;
      }
      // Reload P in A-fragment layout (wave-private patch, no barrier needed).
      const __bf16* pp = &Ps[wave][lrow][akb];
      v16bf pf = cat8(*(const bf16x8*)pp, *(const bf16x8*)(pp + 16));
      // O += P * V  (8 d-tiles of 16)
#pragma unroll
      for (int n = 0; n < 8; ++n) {
        const __bf16* vq = &Vt[n * 16 + lrow][lhi * 8];
        v16bf vf = cat8(*(const bf16x8*)vq, *(const bf16x8*)(vq + 16));
        o[n] = wmma_bf16(pf, vf, o[n]);
      }
    }
    __syncthreads();
  }

#pragma unroll
  for (int n = 0; n < 8; ++n)
#pragma unroll
    for (int r = 0; r < 8; ++r) {
      const int q = q0 + lhi * 8 + r;
      const int dc = n * 16 + lrow;
      O[(size_t)(b * SEQ + q) * EMBED + h * HDIM + dc] =
          (__bf16)(o[n][r] / lsum[r]);
    }
}

// ---------------------------------------------------------------- launch
extern "C" void kernel_launch(void* const* d_in, const int* in_sizes, int n_in,
                              void* d_out, int out_size, void* d_ws,
                              size_t ws_size, hipStream_t stream) {
  const float* x  = (const float*)d_in[0];
  const float* Wq = (const float*)d_in[1];
  const float* Wk = (const float*)d_in[2];
  const float* Wv = (const float*)d_in[3];
  const float* Wo = (const float*)d_in[4];

  char* ws = (char*)d_ws;
  __bf16* Xb  = (__bf16*)(ws);                    // 4096x2048 bf16 (16 MB)
  __bf16* Wqb = (__bf16*)(ws + 16777216);         // 2048x2048 bf16 (8 MB)
  __bf16* Wkb = (__bf16*)(ws + 25165824);
  __bf16* Wvb = (__bf16*)(ws + 33554432);
  __bf16* Wob = (__bf16*)(ws + 41943040);
  __bf16* Qb  = (__bf16*)(ws + 50331648);         // 4096x2048 bf16 (16 MB)
  __bf16* Kb  = (__bf16*)(ws + 67108864);
  __bf16* Vb  = (__bf16*)(ws + 83886080);
  __bf16* Ab  = Xb;  // attention output aliases Xb (dead after QKV GEMMs)

  const int NX = ROWS * EMBED;    // 8,388,608
  const int NW = EMBED * EMBED;   // 4,194,304

  f32_to_bf16_kernel<<<NX / 256, 256, 0, stream>>>(x, Xb, NX);
  f32_to_bf16_kernel<<<NW / 256, 256, 0, stream>>>(Wq, Wqb, NW);
  f32_to_bf16_kernel<<<NW / 256, 256, 0, stream>>>(Wk, Wkb, NW);
  f32_to_bf16_kernel<<<NW / 256, 256, 0, stream>>>(Wv, Wvb, NW);
  f32_to_bf16_kernel<<<NW / 256, 256, 0, stream>>>(Wo, Wob, NW);

  dim3 g(EMBED / 128, ROWS / 128);  // (16, 32)
  gemm_bf16_kernel<__bf16><<<g, 256, 0, stream>>>(Xb, Wqb, Qb, ROWS, EMBED, EMBED);
  gemm_bf16_kernel<__bf16><<<g, 256, 0, stream>>>(Xb, Wkb, Kb, ROWS, EMBED, EMBED);
  gemm_bf16_kernel<__bf16><<<g, 256, 0, stream>>>(Xb, Wvb, Vb, ROWS, EMBED, EMBED);

  rope_inplace_kernel<<<(BATCH * SEQ * HEADS * 64) / 256, 256, 0, stream>>>(Qb, Kb);

  dim3 ga(SEQ / 128, BATCH * HEADS);  // (16, 32)
  flash_attn_kernel<<<ga, 256, 0, stream>>>(Qb, Kb, Vb, Ab);

  gemm_bf16_kernel<float><<<g, 256, 0, stream>>>(Ab, Wob, (float*)d_out,
                                                 ROWS, EMBED, EMBED);
}
